// MultiHeadAttention_67078799229230
// MI455X (gfx1250) — matrix operand order repaired and checked by reading, not compiled, not run on hardware
//
#include <hip/hip_runtime.h>
#include <hip/hip_bf16.h>

#define EMBED 1024
#define HEADS 16
#define HDIM  64
#define SEQ   2048
#define BATCH 2
#define MROWS (BATCH * SEQ) // 4096

typedef __attribute__((ext_vector_type(16))) __bf16 v16bf;
typedef __attribute__((ext_vector_type(8)))  __bf16 v8bf;
typedef __attribute__((ext_vector_type(8)))  float  v8f;
typedef __attribute__((ext_vector_type(4)))  int    v4i;

#if __has_builtin(__builtin_amdgcn_global_load_async_to_lds_b128)
#define USE_ASYNC_LDS 1
#endif

// float -> bf16, round-to-nearest-even (no reliance on native cvt codegen)
static __device__ __forceinline__ __bf16 f2bf(float f) {
  union { float f; unsigned u; } v; v.f = f;
  unsigned r = v.u + 0x7FFFu + ((v.u >> 16) & 1u);
  unsigned short h = (unsigned short)(r >> 16);
  return __builtin_bit_cast(__bf16, h);
}

static __device__ __forceinline__ v8f wmma_bf16(v16bf a, v16bf b, v8f c) {
  // (neg_a, A, neg_b, B, c_mod, C, reuse_a, reuse_b)
  return __builtin_amdgcn_wmma_f32_16x16x32_bf16(false, a, false, b, (short)0, c,
                                                 false, false);
}

// 16-bit A-matrix 16x32 fragment: lane<16 -> row=lane, K {0..15};
// lane>=16 -> row=lane-16, K {16..31}.  rowp points at (row, k0).
static __device__ __forceinline__ v16bf load_a_frag(const __bf16* rowp, int hf) {
  v8bf c0 = *(const v8bf*)(rowp + 16 * hf);
  v8bf c1 = *(const v8bf*)(rowp + 16 * hf + 8);
  v16bf r;
#pragma unroll
  for (int i = 0; i < 8; ++i) { r[i] = c0[i]; r[i + 8] = c1[i]; }
  return r;
}

static __device__ __forceinline__ v16bf assemble16(v8bf c0, v8bf c1) {
  v16bf r;
#pragma unroll
  for (int i = 0; i < 8; ++i) { r[i] = c0[i]; r[i + 8] = c1[i]; }
  return r;
}

// 16-byte global -> LDS copy (async on CDNA5; sync fallback otherwise)
static __device__ __forceinline__ void stage_copy16(const __bf16* g, __bf16* l) {
#ifdef USE_ASYNC_LDS
  typedef __attribute__((address_space(1))) v4i g_v4i;
  typedef __attribute__((address_space(3))) v4i l_v4i;
  __builtin_amdgcn_global_load_async_to_lds_b128((g_v4i*)(void*)g,
                                                 (l_v4i*)(void*)l, 0, 0);
#else
  *(v8bf*)l = *(const v8bf*)g;
#endif
}

static __device__ __forceinline__ void wait_async_pending1() {
#ifdef USE_ASYNC_LDS
  asm volatile("s_wait_asynccnt 0x1" ::: "memory");
#endif
}

// ---------------------------------------------------------------------------
// fp32 -> bf16 elementwise conversion (n multiple of 4)
// ---------------------------------------------------------------------------
__global__ void __launch_bounds__(256)
cvt_f32_bf16(const float* __restrict__ in, __bf16* __restrict__ out, int n) {
  int i = (blockIdx.x * blockDim.x + threadIdx.x) * 4;
  if (i + 3 < n) {
    const float4 f = *(const float4*)(in + i);
    out[i + 0] = f2bf(f.x);
    out[i + 1] = f2bf(f.y);
    out[i + 2] = f2bf(f.z);
    out[i + 3] = f2bf(f.w);
  }
}

// ---------------------------------------------------------------------------
// y = x @ W^T + bias ; x:[4096,1024] bf16, W:[1024,1024] bf16 (row-major)
// mode 0: store bf16 head-split [b][h][l][d]            (Q, K)
// mode 1: store bf16 head-split transposed [b][h][d][l] (V^T)
// mode 2: store fp32 row-major [m][n]                   (final projection)
// Block = 8 waves, all sharing one 64-wide N panel: the W tile (64 x 32 bf16)
// is staged cooperatively into LDS via async global->LDS, double buffered.
// A fragments are register double-buffered.
// grid = (M/16/8) * (N/64) = 32 * 16 = 512 blocks.
// ---------------------------------------------------------------------------
__global__ void __launch_bounds__(256)
gemm_xwT(const __bf16* __restrict__ X, const __bf16* __restrict__ W,
         const float* __restrict__ bias, void* __restrict__ dst, int mode) {
  // 64 rows (n) x 32 k bf16, row stride padded to 40 elems (80B) -> no bank conflicts
  __shared__ __attribute__((aligned(16))) __bf16 ldsW[2][64 * 40];

  const int lane = threadIdx.x & 31;
  const int wave = threadIdx.x >> 5;
  const int ln = lane & 15, hf = lane >> 4;
  const int m0 = ((blockIdx.x & 31) * 8 + wave) * 16;
  const int n0 = (blockIdx.x >> 5) * 64;

  const int srow = threadIdx.x >> 2; // 0..63 : W row within panel
  const int schk = threadIdx.x & 3;  // 0..3  : 16B chunk within 64B k-slice

  v8f zero = {};
  v8f acc[4] = {zero, zero, zero, zero};

  const __bf16* arow = X + (size_t)(m0 + ln) * EMBED;
  const __bf16* wrow = W + (size_t)(n0 + srow) * EMBED + schk * 8;

  // prologue: stage k=0 into buffer 0, prefetch first A fragment
  stage_copy16(wrow, &ldsW[0][srow * 40 + schk * 8]);
  v16bf aCur = load_a_frag(arow, hf);

  int buf = 0;
  for (int k0 = 0; k0 < EMBED; k0 += 32) {
    const int kn = (k0 + 32) & (EMBED - 1); // last iter wraps to 0 (unused)
    // issue next stage + next A fragment before consuming current
    stage_copy16(wrow + kn, &ldsW[buf ^ 1][srow * 40 + schk * 8]);
    v16bf aNext = load_a_frag(arow + kn, hf);

    wait_async_pending1(); // my part of current stage landed
    __syncthreads();       // everyone's part landed

#pragma unroll
    for (int t = 0; t < 4; ++t) {
      const __bf16* lr = &ldsW[buf][(16 * t + ln) * 40 + 16 * hf];
      v16bf b = assemble16(*(const v8bf*)lr, *(const v8bf*)(lr + 8));
      acc[t] = wmma_bf16(aCur, b, acc[t]);
    }
    __syncthreads(); // buffer free before it is refilled next iteration
    buf ^= 1;
    aCur = aNext;
  }

#pragma unroll
  for (int t = 0; t < 4; ++t) {
    const int n = n0 + 16 * t + ln;
    const float bs = bias[n];
#pragma unroll
    for (int r = 0; r < 8; ++r) {
      // C/D f32 layout: lane<16 -> M=r, lane>=16 -> M=r+8 ; N = lane%16
      const int m = m0 + r + 8 * hf;
      const float val = acc[t][r] + bs;
      if (mode == 2) {
        ((float*)dst)[(size_t)m * EMBED + n] = val;
      } else {
        const int b = m >> 11, l = m & (SEQ - 1);
        const int h = n >> 6, d = n & (HDIM - 1);
        const size_t idx =
            (mode == 0) ? (((size_t)(b * HEADS + h) * SEQ + l) * HDIM + d)
                        : (((size_t)(b * HEADS + h) * HDIM + d) * SEQ + l);
        ((__bf16*)dst)[idx] = f2bf(val);
      }
    }
  }
}

// ---------------------------------------------------------------------------
// Flash attention. Q,K: [b][h][l][d] bf16 ; Vt: [b][h][d][l] bf16.
// One wave per 16 query rows, 4 waves/block. 32 keys per iteration.
// K fragments register double-buffered; V fragments issued at loop top so
// their latency hides behind the S-WMMAs and the whole softmax.
// ctx out: bf16 [b][l][h*64+d]  (ready to be A of the O-projection GEMM).
// grid = (N*H) * (2048/64) = 32*32 = 1024 blocks, 128 threads.
// ---------------------------------------------------------------------------
__global__ void __launch_bounds__(128)
attn_flash(const __bf16* __restrict__ Q, const __bf16* __restrict__ K,
           const __bf16* __restrict__ Vt, const int* __restrict__ mask,
           __bf16* __restrict__ ctx) {
  __shared__ __attribute__((aligned(16))) __bf16 pbuf[4][16 * 40]; // 16x32, pad 8

  const int lane = threadIdx.x & 31;
  const int wave = threadIdx.x >> 5;
  const int ln = lane & 15, hf = lane >> 4;
  const int bh = blockIdx.x >> 5;                       // 0..31 = b*16+h
  const int qRow0 = ((blockIdx.x & 31) * 4 + wave) * 16;

  const __bf16* qb = Q + ((size_t)bh * SEQ + qRow0) * HDIM;
  const __bf16* kb = K + (size_t)bh * SEQ * HDIM;
  const __bf16* vtb = Vt + (size_t)bh * HDIM * SEQ;
  const int* mb = mask + (bh >> 4) * SEQ;

  // Q tile 16x64 as two A fragments (d 0..31 / 32..63)
  const v16bf aq0 = load_a_frag(qb + (size_t)ln * HDIM, hf);
  const v16bf aq1 = load_a_frag(qb + (size_t)ln * HDIM + 32, hf);

  v8f zero = {};
  v8f o[4] = {zero, zero, zero, zero}; // 16 x 64 running context
  float rmax[8], rsum[8];
#pragma unroll
  for (int r = 0; r < 8; ++r) { rmax[r] = -3.0e38f; rsum[r] = 0.f; }

  __bf16* pb = &pbuf[wave][0];

  // prologue: K fragments for kk0 = 0
  v16bf kc[4];
  {
    const __bf16* kr0 = kb + (size_t)ln * HDIM;
    const __bf16* kr1 = kb + (size_t)(16 + ln) * HDIM;
    kc[0] = *(const v16bf*)(kr0 + 16 * hf);
    kc[1] = *(const v16bf*)(kr0 + 32 + 16 * hf);
    kc[2] = *(const v16bf*)(kr1 + 16 * hf);
    kc[3] = *(const v16bf*)(kr1 + 32 + 16 * hf);
  }

  for (int kk0 = 0; kk0 < SEQ; kk0 += 32) {
    // ---- V fragments for this iteration: issue now, consume much later
    v16bf vf[4];
#pragma unroll
    for (int t = 0; t < 4; ++t)
      vf[t] = *(const v16bf*)(vtb + (size_t)(16 * t + ln) * SEQ + kk0 + 16 * hf);
    const int mv0 = mb[kk0 + ln];
    const int mv1 = mb[kk0 + 16 + ln];

    // ---- S = Q K^T for two 16-column tiles
    v8f s0 = zero, s1 = zero;
    s0 = wmma_bf16(aq0, kc[0], s0);
    s0 = wmma_bf16(aq1, kc[1], s0);
    s1 = wmma_bf16(aq0, kc[2], s1);
    s1 = wmma_bf16(aq1, kc[3], s1);

    // ---- prefetch next iteration's K fragments (hidden behind softmax)
    const int kn = (kk0 + 32) & (SEQ - 1); // last iter wraps (unused)
    {
      const __bf16* kr0 = kb + (size_t)(kn + ln) * HDIM;
      const __bf16* kr1 = kb + (size_t)(kn + 16 + ln) * HDIM;
      kc[0] = *(const v16bf*)(kr0 + 16 * hf);
      kc[1] = *(const v16bf*)(kr0 + 32 + 16 * hf);
      kc[2] = *(const v16bf*)(kr1 + 16 * hf);
      kc[3] = *(const v16bf*)(kr1 + 32 + 16 * hf);
    }

    float e0[8], e1[8], tm[8];
#pragma unroll
    for (int r = 0; r < 8; ++r) {
      e0[r] = (mv0 != 0 ? s0[r] : -1.0e20f) * 0.03125f; // 1/sqrt(1024)
      e1[r] = (mv1 != 0 ? s1[r] : -1.0e20f) * 0.03125f;
      tm[r] = fmaxf(e0[r], e1[r]);
    }
    // row max across the 16 lanes of each half-wave
#pragma unroll
    for (int off = 1; off < 16; off <<= 1)
#pragma unroll
      for (int r = 0; r < 8; ++r)
        tm[r] = fmaxf(tm[r], __shfl_xor(tm[r], off, 16));

    float corr[8], p0[8], p1[8], ts[8];
#pragma unroll
    for (int r = 0; r < 8; ++r) {
      const float nm = fmaxf(rmax[r], tm[r]);
      corr[r] = __expf(rmax[r] - nm);
      rmax[r] = nm;
      p0[r] = __expf(e0[r] - nm);
      p1[r] = __expf(e1[r] - nm);
      ts[r] = p0[r] + p1[r];
    }
#pragma unroll
    for (int off = 1; off < 16; off <<= 1)
#pragma unroll
      for (int r = 0; r < 8; ++r)
        ts[r] += __shfl_xor(ts[r], off, 16);
#pragma unroll
    for (int r = 0; r < 8; ++r) rsum[r] = rsum[r] * corr[r] + ts[r];
#pragma unroll
    for (int t = 0; t < 4; ++t)
#pragma unroll
      for (int r = 0; r < 8; ++r) o[t][r] *= corr[r];

    // ---- P (C layout) -> LDS -> A-fragment layout
#pragma unroll
    for (int r = 0; r < 8; ++r) {
      pb[(r + 8 * hf) * 40 + ln] = f2bf(p0[r]);
      pb[(r + 8 * hf) * 40 + 16 + ln] = f2bf(p1[r]);
    }
    asm volatile("s_wait_dscnt 0x0" ::: "memory"); // DS in-order per wave; fence compiler

    v16bf pa;
    {
      const __bf16* pr = pb + ln * 40;
      pa = assemble16(*(const v8bf*)(pr + 16 * hf), *(const v8bf*)(pr + 16 * hf + 8));
    }

    // ---- O += P * V  (V fragments issued at loop top, long since resident)
#pragma unroll
    for (int t = 0; t < 4; ++t) o[t] = wmma_bf16(pa, vf[t], o[t]);
    asm volatile("" ::: "memory");
  }

  float inv[8];
#pragma unroll
  for (int r = 0; r < 8; ++r) inv[r] = 1.0f / rsum[r];

  const int b = bh >> 4, h = bh & 15;
#pragma unroll
  for (int t = 0; t < 4; ++t)
#pragma unroll
    for (int r = 0; r < 8; ++r) {
      const int tok = b * SEQ + qRow0 + r + 8 * hf;
      const int col = h * HDIM + 16 * t + ln;
      ctx[(size_t)tok * EMBED + col] = f2bf(o[t][r] * inv[r]);
    }
}

// ---------------------------------------------------------------------------
extern "C" void kernel_launch(void* const* d_in, const int* in_sizes, int n_in,
                              void* d_out, int out_size, void* d_ws, size_t ws_size,
                              hipStream_t stream) {
  (void)in_sizes; (void)n_in; (void)out_size; (void)ws_size;
  const float* values = (const float*)d_in[0];
  const float* key    = (const float*)d_in[1];
  const float* query  = (const float*)d_in[2];
  const int*   mask   = (const int*)d_in[3];
  const float* Wv = (const float*)d_in[4];  const float* bv = (const float*)d_in[5];
  const float* Wk = (const float*)d_in[6];  const float* bk = (const float*)d_in[7];
  const float* Wq = (const float*)d_in[8];  const float* bq = (const float*)d_in[9];
  const float* Wo = (const float*)d_in[10]; const float* bo = (const float*)d_in[11];

  char* ws = (char*)d_ws;
  size_t off = 0;
  auto wsalloc = [&](size_t bytes) {
    void* p = ws + off;
    off += (bytes + 255) & ~(size_t)255;
    return p;
  };
  const size_t XB = (size_t)MROWS * EMBED * sizeof(__bf16); // 8 MiB
  const size_t WB = (size_t)EMBED * EMBED * sizeof(__bf16); // 2 MiB
  __bf16* xv = (__bf16*)wsalloc(XB);
  __bf16* xk = (__bf16*)wsalloc(XB);
  __bf16* xq = (__bf16*)wsalloc(XB);
  __bf16* wv = (__bf16*)wsalloc(WB);
  __bf16* wk = (__bf16*)wsalloc(WB);
  __bf16* wq = (__bf16*)wsalloc(WB);
  __bf16* wo = (__bf16*)wsalloc(WB);
  __bf16* qh = (__bf16*)wsalloc(XB); // [b][h][l][d]
  __bf16* kh = (__bf16*)wsalloc(XB); // [b][h][l][d]
  __bf16* vt = (__bf16*)wsalloc(XB); // [b][h][d][l]
  __bf16* cx = (__bf16*)wsalloc(XB); // [b][l][h*64+d]

  const int NX = MROWS * EMBED;  // 4,194,304
  const int NW = EMBED * EMBED;  // 1,048,576
  cvt_f32_bf16<<<NX / 1024, 256, 0, stream>>>(values, xv, NX);
  cvt_f32_bf16<<<NX / 1024, 256, 0, stream>>>(key,    xk, NX);
  cvt_f32_bf16<<<NX / 1024, 256, 0, stream>>>(query,  xq, NX);
  cvt_f32_bf16<<<NW / 1024, 256, 0, stream>>>(Wv, wv, NW);
  cvt_f32_bf16<<<NW / 1024, 256, 0, stream>>>(Wk, wk, NW);
  cvt_f32_bf16<<<NW / 1024, 256, 0, stream>>>(Wq, wq, NW);
  cvt_f32_bf16<<<NW / 1024, 256, 0, stream>>>(Wo, wo, NW);

  gemm_xwT<<<512, 256, 0, stream>>>(xq, wq, bq, qh, 0);
  gemm_xwT<<<512, 256, 0, stream>>>(xk, wk, bk, kh, 0);
  gemm_xwT<<<512, 256, 0, stream>>>(xv, wv, bv, vt, 1);

  attn_flash<<<1024, 128, 0, stream>>>(qh, kh, vt, mask, cx);

  gemm_xwT<<<512, 256, 0, stream>>>(cx, wo, bo, d_out, 2);
}
